// RepPointRoIExtractor_16518444220755
// MI455X (gfx1250) — matrix operand order repaired and checked by reading, not compiled, not run on hardware
//
#include <hip/hip_runtime.h>
#include <hip/hip_bf16.h>
#include <cstdint>

#define B_ 8
#define P_ 256
#define N_ 1024
#define C_ 256
#define R_ 12
#define M_ (P_*R_)          // 3072 query points per batch
#define NS_ 16
#define RAD2_ 0.04f
#define INV_RADIUS_ 5.0f

typedef __attribute__((ext_vector_type(16))) __bf16 v16bf;
typedef __attribute__((ext_vector_type(8)))  float  v8f;

union Frag16 { v16bf v; float4 f[2]; };

// Flat-aperture rule: for LDS generic pointers, LDS byte offset = addr[31:0].
__device__ __forceinline__ unsigned lds_off(const void* p) {
  return (unsigned)(uintptr_t)p;
}

// A-matrix 16x32 bf16 fragment (contiguous k order used here, same for B): lane holds row (lane&15);
// K = kbase + (lane>=16?16:0) + i, i = 0..15 contiguous.
__device__ __forceinline__ v16bf lda16(const __bf16* base, int row, int stride,
                                       int kbase, int half) {
  Frag16 u;
  const __bf16* p = base + row*stride + kbase + half*16;
  u.f[0] = *(const float4*)(p);
  u.f[1] = *(const float4*)(p + 8);
  return u.v;
}
// B-matrix 32x16 bf16 fragment: lane holds col (lane&15);
// K = kbase + (lane>=16?16:0) + i, i = 0..15 contiguous.
__device__ __forceinline__ v16bf ldb16(const __bf16* wbase, int ncol, int kpad,
                                       int kbase, int half) {
  Frag16 u;
  const __bf16* p = wbase + ncol*kpad + kbase + half*16;
  u.f[0] = *(const float4*)(p);
  u.f[1] = *(const float4*)(p + 8);
  return u.v;
}

// ---------------- kernel 0: seed_features (B,C,N) f32 -> (B,N,C) bf16 ------
__global__ __launch_bounds__(256) void k_tpose(const float* __restrict__ sf,
                                               __bf16* __restrict__ sfT) {
  __shared__ __bf16 tile[32][33];
  int b  = blockIdx.z;
  int n0 = blockIdx.x * 32;
  int c0 = blockIdx.y * 32;
  int tx = threadIdx.x, ty = threadIdx.y;     // 32 x 8
  #pragma unroll
  for (int j = 0; j < 4; ++j) {
    int c = c0 + ty + j*8;
    tile[ty + j*8][tx] = (__bf16)sf[((size_t)b*C_ + c)*N_ + n0 + tx];
  }
  __syncthreads();
  #pragma unroll
  for (int j = 0; j < 4; ++j) {
    int n = n0 + ty + j*8;
    sfT[((size_t)b*N_ + n)*C_ + c0 + tx] = tile[tx][ty + j*8];
  }
}

// ---------------- kernel 1: rep points -------------------------------------
__global__ __launch_bounds__(256) void k_rep(const float* __restrict__ prop,
                                             const float* __restrict__ ref,
                                             const float* __restrict__ dist,
                                             float* __restrict__ rep) {
  int t = blockIdx.x * blockDim.x + threadIdx.x;   // over B*P
  if (t >= B_*P_) return;
  int b = t / P_, p = t - b*P_;
  float ang = prop[t*7 + 6];
  float c = __builtin_cosf(ang), s = __builtin_sinf(ang);
  float rx = ref[t*3+0], ry = ref[t*3+1], rz = ref[t*3+2];
  const float* d = dist + t*6;
  #pragma unroll
  for (int dir = 0; dir < 6; ++dir) {
    float dv = d[dir];
    float lx = 0.f, ly = 0.f, lz = 0.f;
    if (dir == 0) lx =  dv; else if (dir == 1) ly =  dv; else if (dir == 2) lz =  dv;
    else if (dir == 3) lx = -dv; else if (dir == 4) ly = -dv; else lz = -dv;
    #pragma unroll
    for (int si = 0; si < 2; ++si) {
      float sc = (si == 0) ? 1.0f : 0.5f;           // scales = [1.0, 0.5]
      float x = lx*sc, y = ly*sc, z = lz*sc;
      float xr =  x*c + y*s;
      float yr = -x*s + y*c;
      int r = dir*2 + si;
      size_t mi = ((size_t)b*M_ + (size_t)p*R_ + r)*3;
      rep[mi+0] = xr + rx;
      rep[mi+1] = yr + ry;
      rep[mi+2] = z  + rz;
    }
  }
}

// ---------------- kernel 2: ball query (wave per query, ordered first-16) ---
__global__ __launch_bounds__(128) void k_ball(const float* __restrict__ seed_xyz,
                                              const float* __restrict__ rep,
                                              int* __restrict__ idxb) {
  __shared__ int slots[4][NS_];
  int lane = threadIdx.x & 31;
  int wv   = threadIdx.x >> 5;
  int q = blockIdx.x * 4 + wv;
  if (q >= B_*M_) return;
  int b = q / M_;
  float qx = rep[q*3+0], qy = rep[q*3+1], qz = rep[q*3+2];
  if (lane < NS_) slots[wv][lane] = -1;
  asm volatile("s_wait_dscnt 0" ::: "memory");
  const float* sx = seed_xyz + (size_t)b*N_*3;
  int cnt = 0;
  for (int base = 0; base < N_ && cnt < NS_; base += 32) {
    int s = base + lane;
    float dx = sx[s*3+0]-qx, dy = sx[s*3+1]-qy, dz = sx[s*3+2]-qz;
    float d2 = dx*dx + dy*dy + dz*dz;
    int inr = d2 < RAD2_;
    unsigned m32 = (unsigned)__ballot(inr);
    int pre  = __popc(m32 & ((1u << lane) - 1u));
    int slot = cnt + pre;
    if (inr && slot < NS_) slots[wv][slot] = s;
    cnt += __popc(m32);
  }
  asm volatile("s_wait_dscnt 0" ::: "memory");
  if (lane < NS_) {
    int v = slots[wv][lane];
    int first = slots[wv][0];
    int outv = (first < 0) ? 0 : ((v < 0) ? first : v);
    idxb[(size_t)q*NS_ + lane] = outv;
  }
}

// ---------------- kernel 3: async gather + 3-layer WMMA MLP + max ----------
// h layout per row (K=288): [ feat 0..255 | g_xyz 256..258 | zeros 259..287 ]
// w1 columns permuted identically when staged.
__global__ __launch_bounds__(128) void k_mlp(
    const float* __restrict__ seed_xyz, const __bf16* __restrict__ sfT,
    const float* __restrict__ rep, const int* __restrict__ idxb,
    const float* __restrict__ w1, const float* __restrict__ g1, const float* __restrict__ be1,
    const float* __restrict__ w2, const float* __restrict__ g2, const float* __restrict__ be2,
    const float* __restrict__ w3, const float* __restrict__ g3, const float* __restrict__ be3,
    __bf16* __restrict__ featT) {
  extern __shared__ char smem[];
  __bf16* w1s = (__bf16*)smem;           // 128 x 288
  __bf16* w2s = w1s + 128*288;           // 64 x 128
  __bf16* w3s = w2s + 64*128;            // 32 x 64
  float*  scl = (float*)(w3s + 32*64);   // g1,be1,g2,be2,g3,be3 = 448 f32
  __bf16* hb  = (__bf16*)(scl + 448);    // 4 waves x 16 x 288
  __bf16* m1b = hb  + 4*16*288;          // 4 waves x 16 x 128
  __bf16* m2b = m1b + 4*16*128;          // 4 waves x 16 x 64

  int tid = threadIdx.x;
  for (int i = tid; i < 128*288; i += 128) {
    int o = i / 288, c = i - o*288;
    float v;
    if (c < 256)      v = w1[o*259 + 3 + c];      // feature weights
    else if (c < 259) v = w1[o*259 + (c - 256)];  // xyz weights
    else              v = 0.0f;                   // K padding
    w1s[i] = (__bf16)v;
  }
  for (int i = tid; i < 64*128; i += 128) w2s[i] = (__bf16)w2[i];
  for (int i = tid; i < 32*64;  i += 128) w3s[i] = (__bf16)w3[i];
  if (tid < 128) { scl[tid]     = g1[tid]; scl[128+tid] = be1[tid]; }
  if (tid < 64)  { scl[256+tid] = g2[tid]; scl[320+tid] = be2[tid]; }
  if (tid < 32)  { scl[384+tid] = g3[tid]; scl[416+tid] = be3[tid]; }
  __syncthreads();

  int lane = tid & 31;
  int wv   = tid >> 5;
  int half = lane >> 4;
  int col  = lane & 15;
  __bf16* h  = hb  + wv*16*288;
  __bf16* m1 = m1b + wv*16*128;
  __bf16* m2 = m2b + wv*16*64;
  unsigned hbase = lds_off(h);

  int waveId = blockIdx.x*4 + wv;
  int nWaves = gridDim.x*4;

  for (int q = waveId; q < B_*M_; q += nWaves) {
    int b  = q / M_;
    int mq = q - b*M_;
    float qx = rep[q*3+0], qy = rep[q*3+1], qz = rep[q*3+2];
    const int* ids = idxb + (size_t)q*NS_;

    int myid = (lane < 16) ? ids[lane] : 0;
    if (lane < 16) {
      const float* s = seed_xyz + ((size_t)b*N_ + myid)*3;
      h[lane*288+256] = (__bf16)((s[0]-qx)*INV_RADIUS_);
      h[lane*288+257] = (__bf16)((s[1]-qy)*INV_RADIUS_);
      h[lane*288+258] = (__bf16)((s[2]-qz)*INV_RADIUS_);
      #pragma unroll
      for (int c = 259; c < 288; ++c) h[lane*288+c] = (__bf16)0.0f;
    }
    // async copy 16 neighbor rows (256 bf16 each) straight into LDS:
    // per row, each lane moves 16 contiguous bytes, no VGPR round-trip.
    #pragma unroll
    for (int n = 0; n < 16; ++n) {
      int id = __shfl(myid, n, 32);
      const __bf16* g = sfT + ((size_t)b*N_ + id)*C_ + lane*8;
      unsigned ldsa = hbase + (unsigned)((n*288 + lane*8)*2);
      asm volatile("global_load_async_to_lds_b128 %0, %1, off"
                   :: "v"(ldsa), "v"((unsigned long long)(uintptr_t)g)
                   : "memory");
    }
    asm volatile("s_wait_asynccnt 0" ::: "memory");
    asm volatile("s_wait_dscnt 0" ::: "memory");

    // layer 1: (16x288)(288x128)
    #pragma unroll
    for (int nt = 0; nt < 8; ++nt) {
      v8f acc = {};
      #pragma unroll
      for (int ks = 0; ks < 9; ++ks) {
        v16bf a  = lda16(h,   col,          288, ks*32, half);
        v16bf bb = ldb16(w1s, nt*16 + col,  288, ks*32, half);
        acc = __builtin_amdgcn_wmma_f32_16x16x32_bf16(false, a, false, bb,
                                                      (short)0, acc, false, false);
      }
      int o = nt*16 + col;
      float gg = scl[o], bs = scl[128+o];
      #pragma unroll
      for (int v = 0; v < 8; ++v)
        m1[(v + half*8)*128 + o] = (__bf16)fmaxf(acc[v]*gg + bs, 0.0f);
    }
    asm volatile("s_wait_dscnt 0" ::: "memory");

    // layer 2: (16x128)(128x64)
    #pragma unroll
    for (int nt = 0; nt < 4; ++nt) {
      v8f acc = {};
      #pragma unroll
      for (int ks = 0; ks < 4; ++ks) {
        v16bf a  = lda16(m1,  col,         128, ks*32, half);
        v16bf bb = ldb16(w2s, nt*16 + col, 128, ks*32, half);
        acc = __builtin_amdgcn_wmma_f32_16x16x32_bf16(false, a, false, bb,
                                                      (short)0, acc, false, false);
      }
      int o = nt*16 + col;
      float gg = scl[256+o], bs = scl[320+o];
      #pragma unroll
      for (int v = 0; v < 8; ++v)
        m2[(v + half*8)*64 + o] = (__bf16)fmaxf(acc[v]*gg + bs, 0.0f);
    }
    asm volatile("s_wait_dscnt 0" ::: "memory");

    // layer 3: (16x64)(64x32), then max over the 16 neighbor rows
    float cmax[2];
    #pragma unroll
    for (int nt = 0; nt < 2; ++nt) {
      v8f acc = {};
      #pragma unroll
      for (int ks = 0; ks < 2; ++ks) {
        v16bf a  = lda16(m2,  col,         64, ks*32, half);
        v16bf bb = ldb16(w3s, nt*16 + col, 64, ks*32, half);
        acc = __builtin_amdgcn_wmma_f32_16x16x32_bf16(false, a, false, bb,
                                                      (short)0, acc, false, false);
      }
      int o = nt*16 + col;
      float gg = scl[384+o], bs = scl[416+o];
      float mv = 0.0f;                       // post-ReLU values are >= 0
      #pragma unroll
      for (int v = 0; v < 8; ++v) mv = fmaxf(mv, fmaxf(acc[v]*gg + bs, 0.0f));
      mv = fmaxf(mv, __shfl_xor(mv, 16, 32));   // merge rows 0..7 with 8..15
      cmax[nt] = mv;
    }
    if (lane < 16) {
      int p = mq / R_;
      int r = mq - p*R_;
      __bf16* fT = featT + ((size_t)b*P_ + p)*384;   // featT[b][p][c], c = o*R + r
      fT[col*R_ + r]        = (__bf16)cmax[0];
      fT[(16 + col)*R_ + r] = (__bf16)cmax[1];
    }
  }
}

// ---------------- kernel 4: out = wr(128x384) @ featT[b]^T + br ------------
__global__ __launch_bounds__(256) void k_final(const float* __restrict__ wr,
                                               const float* __restrict__ br,
                                               const __bf16* __restrict__ featT,
                                               float* __restrict__ out) {
  extern __shared__ char smem[];
  __bf16* wrs = (__bf16*)smem;            // 128 x 384
  float*  brs = (float*)(wrs + 128*384);
  int tid = threadIdx.x;
  for (int i = tid; i < 128*384; i += 256) wrs[i] = (__bf16)wr[i];
  if (tid < 128) brs[tid] = br[tid];
  __syncthreads();

  int b    = blockIdx.x;
  int lane = tid & 31, wv = tid >> 5;
  int half = lane >> 4, col = lane & 15;
  int obase = wv * 16;                     // 8 waves -> 8 M-tiles of 16 channels

  v16bf af[12];
  #pragma unroll
  for (int ks = 0; ks < 12; ++ks)
    af[ks] = lda16(wrs + obase*384, col, 384, ks*32, half);

  const __bf16* fb = featT + (size_t)b*P_*384;
  for (int pt = 0; pt < 16; ++pt) {
    v8f acc = {};
    #pragma unroll
    for (int ks = 0; ks < 12; ++ks) {
      Frag16 u;                            // B[k,n] = featT[b][pt*16+n][k]
      const __bf16* p = fb + (size_t)(pt*16 + col)*384 + ks*32 + half*16;
      u.f[0] = *(const float4*)(p);
      u.f[1] = *(const float4*)(p + 8);
      acc = __builtin_amdgcn_wmma_f32_16x16x32_bf16(false, af[ks], false, u.v,
                                                    (short)0, acc, false, false);
    }
    #pragma unroll
    for (int v = 0; v < 8; ++v) {
      int o = obase + v + half*8;
      out[((size_t)b*128 + o)*P_ + pt*16 + col] = acc[v] + brs[o];
    }
  }
}

extern "C" void kernel_launch(void* const* d_in, const int* in_sizes, int n_in,
                              void* d_out, int out_size, void* d_ws, size_t ws_size,
                              hipStream_t stream) {
  (void)in_sizes; (void)n_in; (void)out_size; (void)ws_size;
  const float* seed_xyz      = (const float*)d_in[0];
  const float* seed_features = (const float*)d_in[1];
  const float* proposals     = (const float*)d_in[2];
  const float* ref_points    = (const float*)d_in[3];
  const float* distance      = (const float*)d_in[4];
  const float* w1  = (const float*)d_in[5];
  const float* g1  = (const float*)d_in[6];
  const float* be1 = (const float*)d_in[7];
  const float* w2  = (const float*)d_in[8];
  const float* g2  = (const float*)d_in[9];
  const float* be2 = (const float*)d_in[10];
  const float* w3  = (const float*)d_in[11];
  const float* g3  = (const float*)d_in[12];
  const float* be3 = (const float*)d_in[13];
  const float* wr  = (const float*)d_in[14];
  const float* br  = (const float*)d_in[15];
  float* out = (float*)d_out;

  char* ws = (char*)d_ws;
  size_t off = 0;
  float*  rep   = (float*)(ws + off);  off += (size_t)B_*M_*3*4;     // 294912
  int*    idxb  = (int*)  (ws + off);  off += (size_t)B_*M_*NS_*4;   // 1572864
  __bf16* featT = (__bf16*)(ws + off); off += (size_t)B_*P_*384*2;   // 1572864
  __bf16* sfT   = (__bf16*)(ws + off);                               // B*N*C*2 = 4 MB

  k_tpose<<<dim3(N_/32, C_/32, B_), dim3(32, 8), 0, stream>>>(seed_features, sfT);
  k_rep <<<(B_*P_ + 255)/256, 256, 0, stream>>>(proposals, ref_points, distance, rep);
  k_ball<<<(B_*M_)/4, 128, 0, stream>>>(seed_xyz, rep, idxb);

  size_t smem3 = (size_t)(128*288 + 64*128 + 32*64)*2 + 448*4
               + (size_t)4*16*(288 + 128 + 64)*2;                    // 157440 B
  k_mlp<<<128, 128, smem3, stream>>>(seed_xyz, sfT, rep, idxb,
                                     w1, g1, be1, w2, g2, be2, w3, g3, be3, featT);

  size_t smem4 = (size_t)128*384*2 + 128*4;                          // 98816 B
  k_final<<<B_, 256, smem4, stream>>>(wr, br, featT, out);
}
